// SequenceEncoder_68942815035944
// MI455X (gfx1250) — matrix operand order repaired and checked by reading, not compiled, not run on hardware
//
#include <hip/hip_runtime.h>
#include <math.h>

typedef __attribute__((ext_vector_type(16))) _Float16 v16h;
typedef __attribute__((ext_vector_type(8)))  _Float16 v8h;
typedef __attribute__((ext_vector_type(8)))  float    v8f;

#define BB      4
#define LL      1024
#define DMODEL  256
#define NLAYERS 4
#define DSTATE  16
#define DCONV   4
#define DINNER  512
#define DTRANK  16
#define USEDC   224
#define MROWS   (BB * LL)   // 4096

// ---------------------------------------------------------------------------
// Weight pack: f32 W[N,K] -> f16 Wp[Np,Kp] zero-padded (Np,Kp multiples of 64)
// ---------------------------------------------------------------------------
__global__ void pack_w_kernel(const float* __restrict__ W, int N, int K,
                              _Float16* __restrict__ Wp, int Np, int Kp)
{
    int gid = blockIdx.x * blockDim.x + threadIdx.x;
    if (gid >= Np * Kp) return;
    int n = gid / Kp, k = gid % Kp;
    float v = (n < N && k < K) ? W[(size_t)n * K + k] : 0.0f;
    Wp[gid] = (_Float16)v;
}

// ---------------------------------------------------------------------------
// WMMA GEMM: out[M,N] = act( A_f16[M,K] * Wp_f16[N,K]^T + bias )
//   All dims pre-padded: K % 64 == 0, N % 64 == 0, lda == K. No guards.
//   act: 0 = none, 1 = softplus.  accum: 1 = out += result.
// Block = 256 threads (8 wave32s), block tile 32(M) x 64(N), K step 64
// (two 32-k slabs -> two v_wmma per step). Double-buffered LDS, software-
// pipelined: next tile's global_load_b128s issue before current WMMAs,
// one barrier per k-step.
// ---------------------------------------------------------------------------
#define ASTR 72   // halves per As row (64 k + 8 pad) = 144 B (16B multiple)
#define BSTR 72

__global__ __launch_bounds__(256)
void gemm_wmma_kernel(const _Float16* __restrict__ A, int K,
                      const _Float16* __restrict__ Wp, int N,
                      const float* __restrict__ bias,
                      float* __restrict__ out, int act, int accum)
{
    __shared__ __align__(16) _Float16 As[2][32 * ASTR];
    __shared__ __align__(16) _Float16 Bs[2][64 * BSTR];

    const int tid  = threadIdx.x;
    const int lane = tid & 31;
    const int wave = tid >> 5;
    const int wm   = wave & 1;        // wave M index (0..1)
    const int wn   = wave >> 1;       // wave N index (0..3)
    const int half = lane >> 4;
    const int ml   = lane & 15;
    const int m0   = blockIdx.x * 32;
    const int n0   = blockIdx.y * 64;

    // cooperative staging assignments (all aligned 16B transfers)
    const int ar = tid >> 3;              // A row 0..31, 8 halves each
    const int ak = (tid & 7) * 8;
    const int bn = tid >> 2;              // W n-row 0..63, 16 halves each
    const int bk = (tid & 3) * 16;

    const _Float16* Ag = &A[(size_t)(m0 + ar) * K + ak];
    const _Float16* Wg = &Wp[(size_t)(n0 + bn) * K + bk];
    const int aoff = ar * ASTR + ak;
    const int boff = bn * BSTR + bk;

    // prologue: fetch + stage k-step 0 into buffer 0
    v8h ta  = *reinterpret_cast<const v8h*>(Ag);
    v8h tb0 = *reinterpret_cast<const v8h*>(Wg);
    v8h tb1 = *reinterpret_cast<const v8h*>(Wg + 8);
    *reinterpret_cast<v8h*>(&As[0][aoff])     = ta;
    *reinterpret_cast<v8h*>(&Bs[0][boff])     = tb0;
    *reinterpret_cast<v8h*>(&Bs[0][boff + 8]) = tb1;
    __syncthreads();

    v8f c = {};
    const int ksteps = K >> 6;
    for (int ks = 0; ks < ksteps; ++ks) {
        const int  cur  = ks & 1;
        const int  nxt  = cur ^ 1;
        const bool more = (ks + 1) < ksteps;
        // issue next tile's global loads early (overlap with WMMA below)
        if (more) {
            const int k1 = (ks + 1) << 6;
            ta  = *reinterpret_cast<const v8h*>(Ag + k1);
            tb0 = *reinterpret_cast<const v8h*>(Wg + k1);
            tb1 = *reinterpret_cast<const v8h*>(Wg + k1 + 8);
        }
        // consume current buffer: two 32-k slabs
#pragma unroll
        for (int j = 0; j < 2; ++j) {
            const _Float16* ap = &As[cur][(wm * 16 + ml) * ASTR + j * 32 + half * 8];
            v8h alo = *reinterpret_cast<const v8h*>(ap);
            v8h ahi = *reinterpret_cast<const v8h*>(ap + 16);
            const _Float16* bp = &Bs[cur][(wn * 16 + ml) * BSTR + j * 32 + half * 8];
            v8h blo = *reinterpret_cast<const v8h*>(bp);
            v8h bhi = *reinterpret_cast<const v8h*>(bp + 16);
            v16h a  = __builtin_shufflevector(alo, ahi, 0,1,2,3,4,5,6,7,8,9,10,11,12,13,14,15);
            v16h bf = __builtin_shufflevector(blo, bhi, 0,1,2,3,4,5,6,7,8,9,10,11,12,13,14,15);
            c = __builtin_amdgcn_wmma_f32_16x16x32_f16(false, a, false, bf,
                                                       (short)0, c, false, false);
        }
        // stage next tile into the other buffer; single barrier per k-step
        if (more) {
            *reinterpret_cast<v8h*>(&As[nxt][aoff])     = ta;
            *reinterpret_cast<v8h*>(&Bs[nxt][boff])     = tb0;
            *reinterpret_cast<v8h*>(&Bs[nxt][boff + 8]) = tb1;
            __syncthreads();
        }
    }

    // C/D layout: VGPR r -> M = half*8 + r, N = lane&15
    const int ng = n0 + wn * 16 + ml;
    float bv = bias ? bias[ng] : 0.0f;
#pragma unroll
    for (int r = 0; r < 8; ++r) {
        int   mg = m0 + wm * 16 + half * 8 + r;
        float v  = c[r] + bv;
        if (act == 1) v = (v > 20.0f) ? v : log1pf(expf(v)); // softplus
        size_t o = (size_t)mg * N + ng;
        if (accum) out[o] += v; else out[o] = v;
    }
}

// ---------------------------------------------------------------------------
// Token embedding gather -> f16 matrix E[MROWS, 256] (cols 224..255 zero)
// ---------------------------------------------------------------------------
__global__ void embed_gather_kernel(const float* __restrict__ hist,
                                    const float* __restrict__ img_e,
                                    const float* __restrict__ saki_e,
                                    const float* __restrict__ act_e,
                                    const float* __restrict__ lab_e,
                                    _Float16* __restrict__ E)
{
    int gid = blockIdx.x * blockDim.x + threadIdx.x;
    if (gid >= MROWS * 256) return;
    int row = gid >> 8, u = gid & 255;
    const float* h = hist + (size_t)row * 5;
    float v = 0.0f;
    if (u < 64) {
        int idx = (int)fmaxf(h[0], 0.0f);
        v = img_e[(size_t)idx * 64 + u];
    } else if (u < 96) {
        int s = (int)fmaxf(h[1], 0.0f);
        v = saki_e[s * 32 + (u - 64)];
    } else if (u < 160) {
        int a = (int)fmaxf(h[2], 0.0f);
        v = act_e[a * 64 + (u - 96)];
    } else if (u < 224) {
        int lv  = (int)fmaxf(h[4], 0.0f);
        int lid = (h[3] > 0.5f) ? lv : 2;
        v = lab_e[lid * 64 + (u - 160)];
    }
    E[gid] = (_Float16)v;
}

// ---------------------------------------------------------------------------
// LayerNorm over 256 cols; one row per block of 256 threads.
// ---------------------------------------------------------------------------
__global__ __launch_bounds__(256)
void ln_to_f16_kernel(const float* __restrict__ x, const float* __restrict__ g,
                      const float* __restrict__ b, _Float16* __restrict__ out)
{
    __shared__ float red[256];
    int row = blockIdx.x, c = threadIdx.x;
    float v = x[(size_t)row * 256 + c];
    red[c] = v; __syncthreads();
    for (int off = 128; off > 0; off >>= 1) { if (c < off) red[c] += red[c + off]; __syncthreads(); }
    float mean = red[0] * (1.0f / 256.0f);
    __syncthreads();
    float dv = v - mean;
    red[c] = dv * dv; __syncthreads();
    for (int off = 128; off > 0; off >>= 1) { if (c < off) red[c] += red[c + off]; __syncthreads(); }
    float var = red[0] * (1.0f / 256.0f);
    float rs  = rsqrtf(var + 1e-5f);
    out[(size_t)row * 256 + c] = (_Float16)(dv * rs * g[c] + b[c]);
}

__global__ __launch_bounds__(256)
void ln_to_f32_kernel(const float* __restrict__ x, const float* __restrict__ g,
                      const float* __restrict__ b, float* __restrict__ out)
{
    __shared__ float red[256];
    int row = blockIdx.x, c = threadIdx.x;
    float v = x[(size_t)row * 256 + c];
    red[c] = v; __syncthreads();
    for (int off = 128; off > 0; off >>= 1) { if (c < off) red[c] += red[c + off]; __syncthreads(); }
    float mean = red[0] * (1.0f / 256.0f);
    __syncthreads();
    float dv = v - mean;
    red[c] = dv * dv; __syncthreads();
    for (int off = 128; off > 0; off >>= 1) { if (c < off) red[c] += red[c + off]; __syncthreads(); }
    float var = red[0] * (1.0f / 256.0f);
    float rs  = rsqrtf(var + 1e-5f);
    out[(size_t)row * 256 + c] = dv * rs * g[c] + b[c];
}

// ---------------------------------------------------------------------------
// Depthwise causal conv (kernel 4) over xi = xz[:, :512], + bias, SiLU.
// ---------------------------------------------------------------------------
__global__ void conv_silu_kernel(const float* __restrict__ xz,
                                 const float* __restrict__ cw,
                                 const float* __restrict__ cb,
                                 float* __restrict__ xc, _Float16* __restrict__ xch)
{
    int gid = blockIdx.x * blockDim.x + threadIdx.x;
    if (gid >= MROWS * DINNER) return;
    int d = gid % DINNER;
    int t = (gid / DINNER) % LL;
    int b = gid / (DINNER * LL);
    float acc = cb[d];
#pragma unroll
    for (int j = 0; j < 4; ++j) {
        int   tt = t - 3 + j;
        int   tc = (tt < 0) ? 0 : tt;
        float mk = (tt < 0) ? 0.0f : 1.0f;
        acc += cw[d * 4 + j] * mk * xz[(size_t)(b * LL + tc) * 1024 + d];
    }
    float s = acc / (1.0f + expf(-acc));   // SiLU
    xc[gid]  = s;
    xch[gid] = (_Float16)s;
}

// dbc[:, :16] (stride 64) -> f16 dt matrix [MROWS, 64] (cols 16..63 zero)
__global__ void dt_extract_kernel(const float* __restrict__ dbc, _Float16* __restrict__ dth)
{
    int gid = blockIdx.x * blockDim.x + threadIdx.x;
    if (gid >= MROWS * 64) return;
    int row = gid >> 6, u = gid & 63;
    float v = (u < DTRANK) ? dbc[(size_t)row * 64 + u] : 0.0f;
    dth[gid] = (_Float16)v;
}

// ---------------------------------------------------------------------------
// Selective scan: one (b,d) channel per 16-lane group, one state s per lane.
// h_s(t) = exp(delta*A_s) h_s + delta*B_s*xc ; y = sum_s h_s*C_s + xc*D
// Fused with SiLU(z) gate, writes f16 A-input for out_proj GEMM.
// dbc row stride is 64: [0:16]=dt, [16:32]=B, [32:48]=C.
// ---------------------------------------------------------------------------
__global__ __launch_bounds__(256)
void scan_kernel(const float* __restrict__ delta, const float* __restrict__ dbc,
                 const float* __restrict__ xc, const float* __restrict__ xz,
                 const float* __restrict__ A_log, const float* __restrict__ Dp,
                 _Float16* __restrict__ yg)
{
    int tid = blockIdx.x * 256 + threadIdx.x;
    int grp = tid >> 4;                 // channel 0 .. B*DINNER-1
    int s   = tid & 15;                 // state index
    int b   = grp >> 9;                 // / DINNER
    int d   = grp & (DINNER - 1);
    float A  = -expf(A_log[d * DSTATE + s]);
    float Dd = Dp[d];
    float h  = 0.0f;
    for (int t = 0; t < LL; ++t) {
        size_t row = (size_t)(b * LL + t);
        float dv = delta[row * DINNER + d];
        float xv = xc[row * DINNER + d];
        float Bm = dbc[row * 64 + 16 + s];
        float Cm = dbc[row * 64 + 32 + s];
        h = expf(dv * A) * h + dv * Bm * xv;
        float p = h * Cm;
        p += __shfl_xor(p, 1, 32);
        p += __shfl_xor(p, 2, 32);
        p += __shfl_xor(p, 4, 32);
        p += __shfl_xor(p, 8, 32);      // sum over 16 state lanes
        if (s == 0) {
            float y  = p + xv * Dd;
            float z  = xz[row * 1024 + DINNER + d];
            float gz = z / (1.0f + expf(-z));      // SiLU gate
            yg[row * DINNER + d] = (_Float16)(y * gz);
        }
    }
}

// ---------------------------------------------------------------------------
extern "C" void kernel_launch(void* const* d_in, const int* in_sizes, int n_in,
                              void* d_out, int out_size, void* d_ws, size_t ws_size,
                              hipStream_t stream)
{
    const float* history     = (const float*)d_in[0];
    const float* image_emb   = (const float*)d_in[1];
    const float* saki_emb    = (const float*)d_in[2];
    const float* action_emb  = (const float*)d_in[3];
    const float* label_emb   = (const float*)d_in[4];
    const float* token_pad_w = (const float*)d_in[5];
    const float* token_pad_b = (const float*)d_in[6];
    const float* ln_g        = (const float*)d_in[7];
    const float* ln_b        = (const float*)d_in[8];
    const float* in_proj_w   = (const float*)d_in[9];
    const float* conv_w      = (const float*)d_in[10];
    const float* conv_b      = (const float*)d_in[11];
    const float* x_proj_w    = (const float*)d_in[12];
    const float* dt_proj_w   = (const float*)d_in[13];
    const float* dt_proj_b   = (const float*)d_in[14];
    const float* A_log       = (const float*)d_in[15];
    const float* Dp          = (const float*)d_in[16];
    const float* out_proj_w  = (const float*)d_in[17];
    const float* fn_g        = (const float*)d_in[18];
    const float* fn_b        = (const float*)d_in[19];

    // workspace carve-out
    char* wsp = (char*)d_ws;
    auto carve = [&](size_t bytes) -> void* {
        void* p = (void*)wsp;
        wsp += (bytes + 255) & ~(size_t)255;
        return p;
    };
    // activations
    float*    tok   = (float*)   carve((size_t)MROWS * 256 * 4);
    _Float16* E     = (_Float16*)carve((size_t)MROWS * 256 * 2);
    _Float16* xln   = (_Float16*)carve((size_t)MROWS * 256 * 2);
    float*    xz    = (float*)   carve((size_t)MROWS * 1024 * 4);
    float*    xc    = (float*)   carve((size_t)MROWS * DINNER * 4);
    _Float16* xch   = (_Float16*)carve((size_t)MROWS * DINNER * 2);
    float*    dbc   = (float*)   carve((size_t)MROWS * 64 * 4);
    _Float16* dth   = (_Float16*)carve((size_t)MROWS * 64 * 2);
    float*    delta = (float*)   carve((size_t)MROWS * DINNER * 4);
    _Float16* yg    = (_Float16*)carve((size_t)MROWS * DINNER * 2);
    // packed f16 weights (zero-padded to multiples of 64)
    _Float16* Wtok  = (_Float16*)carve((size_t)256 * 256 * 2);
    _Float16* Win   = (_Float16*)carve((size_t)NLAYERS * 1024 * 256 * 2);
    _Float16* Wx    = (_Float16*)carve((size_t)NLAYERS * 64 * 512 * 2);
    _Float16* Wdt   = (_Float16*)carve((size_t)NLAYERS * 512 * 64 * 2);
    _Float16* Wout  = (_Float16*)carve((size_t)NLAYERS * 256 * 512 * 2);

    // 0) pack all weights to padded f16 (tiny; lives in L2)
    pack_w_kernel<<<(256 * 256 + 255) / 256, 256, 0, stream>>>(
        token_pad_w, 256, USEDC, Wtok, 256, 256);
    for (int i = 0; i < NLAYERS; ++i) {
        pack_w_kernel<<<(1024 * 256 + 255) / 256, 256, 0, stream>>>(
            in_proj_w + (size_t)i * 1024 * 256, 1024, 256, Win + (size_t)i * 1024 * 256, 1024, 256);
        pack_w_kernel<<<(64 * 512 + 255) / 256, 256, 0, stream>>>(
            x_proj_w + (size_t)i * 48 * 512, 48, 512, Wx + (size_t)i * 64 * 512, 64, 512);
        pack_w_kernel<<<(512 * 64 + 255) / 256, 256, 0, stream>>>(
            dt_proj_w + (size_t)i * 512 * 16, 512, 16, Wdt + (size_t)i * 512 * 64, 512, 64);
        pack_w_kernel<<<(256 * 512 + 255) / 256, 256, 0, stream>>>(
            out_proj_w + (size_t)i * 256 * 512, 256, 512, Wout + (size_t)i * 256 * 512, 256, 512);
    }

    // 1) embed + token pad projection: tok = E * Wtok^T + b   (K=256 padded)
    embed_gather_kernel<<<(MROWS * 256 + 255) / 256, 256, 0, stream>>>(
        history, image_emb, saki_emb, action_emb, label_emb, E);
    gemm_wmma_kernel<<<dim3(MROWS / 32, 256 / 64), 256, 0, stream>>>(
        E, 256, Wtok, 256, token_pad_b, tok, 0, 0);

    // 2) mamba layers
    for (int i = 0; i < NLAYERS; ++i) {
        ln_to_f16_kernel<<<MROWS, 256, 0, stream>>>(tok, ln_g + i * 256, ln_b + i * 256, xln);

        // xz = xln * in_proj^T   (M=4096, N=1024, K=256)
        gemm_wmma_kernel<<<dim3(MROWS / 32, 1024 / 64), 256, 0, stream>>>(
            xln, 256, Win + (size_t)i * 1024 * 256, 1024, nullptr, xz, 0, 0);

        conv_silu_kernel<<<(MROWS * DINNER + 255) / 256, 256, 0, stream>>>(
            xz, conv_w + (size_t)i * DINNER * 4, conv_b + i * DINNER, xc, xch);

        // dbc = xc * x_proj^T    (N=64 padded, K=512); padded cols are exact 0
        gemm_wmma_kernel<<<dim3(MROWS / 32, 64 / 64), 256, 0, stream>>>(
            xch, DINNER, Wx + (size_t)i * 64 * 512, 64, nullptr, dbc, 0, 0);

        dt_extract_kernel<<<(MROWS * 64 + 255) / 256, 256, 0, stream>>>(dbc, dth);

        // delta = softplus(dt * dt_proj^T + dt_b)  (N=512, K=64 padded)
        gemm_wmma_kernel<<<dim3(MROWS / 32, DINNER / 64), 256, 0, stream>>>(
            dth, 64, Wdt + (size_t)i * 512 * 64, DINNER, dt_proj_b + i * DINNER, delta, 1, 0);

        // selective scan + D-skip + SiLU(z) gate -> yg (f16)
        scan_kernel<<<(BB * DINNER * DSTATE) / 256, 256, 0, stream>>>(
            delta, dbc, xc, xz, A_log + (size_t)i * DINNER * DSTATE, Dp + i * DINNER, yg);

        // tok += yg * out_proj^T  (N=256, K=512, residual accumulate)
        gemm_wmma_kernel<<<dim3(MROWS / 32, 256 / 64), 256, 0, stream>>>(
            yg, DINNER, Wout + (size_t)i * 256 * 512, 256, nullptr, tok, 0, 1);
    }

    // 3) final layernorm -> d_out (f32)
    ln_to_f32_kernel<<<MROWS, 256, 0, stream>>>(tok, fn_g, fn_b, (float*)d_out);
}
